// HierarchicalRC_56341380989597
// MI455X (gfx1250) — compile-verified, matrix operand order checked by reading, not run on hardware
//
#include <hip/hip_runtime.h>
#include <cstdint>

// CDNA5 / gfx1250 WMMA operand types (wave32)
typedef __attribute__((ext_vector_type(16))) __bf16 v16bf;
typedef __attribute__((ext_vector_type(8)))  float  v8f;

#define N_NODES 8192
#define D_FEAT  256
#define TILE_M  128
#define TILE_N  64
#define TILE_K  32

__device__ __forceinline__ float sigmoidf_(float x) { return 1.0f / (1.0f + __expf(-x)); }

__device__ __forceinline__ v8f wmma_bf16(v16bf a, v16bf b, v8f c) {
    // (neg_a, A, neg_b, B, c_mod, C, reuse_a, reuse_b)
    return __builtin_amdgcn_wmma_f32_16x16x32_bf16(false, a, false, b, (short)0, c,
                                                   false, false);
}

// gfx1250 async copy global->LDS, 16B per lane, tracked by ASYNCcnt.
// Low 32 bits of a flat LDS pointer are the LDS byte offset (ISA 10.2 aperture rule).
__device__ __forceinline__ void async_load_b128(const void* gptr, void* lds_ptr) {
    unsigned long long ga = (unsigned long long)(uintptr_t)gptr;
    unsigned int       la = (unsigned int)(uintptr_t)lds_ptr;
    asm volatile("global_load_async_to_lds_b128 %0, %1, off" :: "v"(la), "v"(ga) : "memory");
}
__device__ __forceinline__ void wait_async0() {
    asm volatile("s_wait_asynccnt 0x0" ::: "memory");
}

// ---- one-time pre-pass: f32 src[K][N] -> bf16 hi/lo dst[N][K] (transpose + split) ----
__global__ __launch_bounds__(256) void transpose_split(
    const float* __restrict__ src, int K, int N,
    __bf16* __restrict__ dhi, __bf16* __restrict__ dlo)
{
    __shared__ float tile[32][33];
    const int bk = blockIdx.x * 32;
    const int bn = blockIdx.y * 32;
    const int tx = threadIdx.x & 31;
    const int ty = threadIdx.x >> 5;  // 0..7
    #pragma unroll
    for (int i = 0; i < 4; ++i)
        tile[ty + i * 8][tx] = src[(size_t)(bk + ty + i * 8) * N + bn + tx];
    __syncthreads();
    #pragma unroll
    for (int i = 0; i < 4; ++i) {
        const int n = bn + ty + i * 8;
        const float f = tile[tx][ty + i * 8];
        const __bf16 h = (__bf16)f;
        dhi[(size_t)n * K + bk + tx] = h;
        dlo[(size_t)n * K + bk + tx] = (__bf16)(f - (float)h);
    }
}

// C[M,N=256] = A[M,K](f32,lda) @ B[K,N], with B pre-split/transposed to bf16 Bt[N][K].
// A is split to bf16 hi/lo on the fly; 3 WMMAs per K-slice (bf16x3 ~ f32 accuracy).
// mode 0: Out = C.   mode 1: Out = s*relu(C) + (1-s)*Xres, s = sig(wg)*sig(wl[row]).
__global__ __launch_bounds__(256) void gcn_gemm_bf16x3(
    const float* __restrict__ A, int lda,
    const __bf16* __restrict__ Bthi, const __bf16* __restrict__ Btlo,
    int K, int mode,
    const float* __restrict__ Xres,
    const float* __restrict__ w_local,
    const float* __restrict__ w_global,
    float* __restrict__ Out, int ldo)
{
    __shared__ __bf16 sAhi[2][TILE_M][TILE_K];
    __shared__ __bf16 sAlo[2][TILE_M][TILE_K];
    __shared__ __bf16 sBhi[2][TILE_N][TILE_K];  // [n][k], matches Bt layout
    __shared__ __bf16 sBlo[2][TILE_N][TILE_K];

    const int t     = threadIdx.x;
    const int lane  = t & 31;
    const int wave  = t >> 5;
    const int mwave = wave & 3;   // 4 waves tile M (32 rows each)
    const int nwave = wave >> 2;  // 2 waves tile N (32 cols each)
    const int rowA0 = blockIdx.y * TILE_M;
    const int colB0 = blockIdx.x * TILE_N;

    const int fr = lane & 15;         // fragment row/col within 16
    const int fk = (lane >> 4) * 16;  // half-wave K split

    // B async staging coords: 256 threads cover 64 rows x (32*2B = 64B) as 4x16B chunks
    const int bn  = t >> 2;           // 0..63
    const int bkc = (t & 3) * 8;      // bf16 element offset: 0,8,16,24

    float4 areg[4];

    auto loadA = [&](int k0) {
        #pragma unroll
        for (int i = 0; i < 4; ++i) {
            const int slot = t * 4 + i;       // 1024 float4 slots over 128x32
            const int r = slot >> 3;
            const int c = (slot & 7) * 4;
            areg[i] = *(const float4*)(A + (size_t)(rowA0 + r) * lda + k0 + c);
        }
    };
    auto storeA = [&](int buf) {
        #pragma unroll
        for (int i = 0; i < 4; ++i) {
            const int slot = t * 4 + i;
            const int r = slot >> 3;
            const int c = (slot & 7) * 4;
            const float4 v = areg[i];
            __attribute__((aligned(8))) __bf16 h4[4], l4[4];
            h4[0] = (__bf16)v.x;  l4[0] = (__bf16)(v.x - (float)h4[0]);
            h4[1] = (__bf16)v.y;  l4[1] = (__bf16)(v.y - (float)h4[1]);
            h4[2] = (__bf16)v.z;  l4[2] = (__bf16)(v.z - (float)h4[2]);
            h4[3] = (__bf16)v.w;  l4[3] = (__bf16)(v.w - (float)h4[3]);
            *(uint2*)&sAhi[buf][r][c] = *(const uint2*)h4;   // ds_store_b64
            *(uint2*)&sAlo[buf][r][c] = *(const uint2*)l4;   // ds_store_b64
        }
    };
    auto stageB = [&](int buf, int k0) {
        const size_t go = (size_t)(colB0 + bn) * (size_t)K + (size_t)(k0 + bkc);
        async_load_b128(Bthi + go, &sBhi[buf][bn][bkc]);
        async_load_b128(Btlo + go, &sBlo[buf][bn][bkc]);
    };

    v8f acc[2][2] = {};

    // ---- prologue: stage tile 0 ----
    stageB(0, 0);
    loadA(0);
    storeA(0);
    wait_async0();
    __syncthreads();

    int cur = 0;
    for (int k0 = 0; k0 < K; k0 += TILE_K) {
        const int nxt = cur ^ 1;
        const bool more = (k0 + TILE_K) < K;
        if (more) {
            stageB(nxt, k0 + TILE_K);   // fire async B copy for next tile
            loadA(k0 + TILE_K);         // fire global A loads for next tile
        }

        // ---- fragments from current buffer (2 x ds_load_b128 each) ----
        v16bf ahi[2], alo[2], bhi[2], blo[2];
        #pragma unroll
        for (int mi = 0; mi < 2; ++mi) {
            const int r = mwave * 32 + mi * 16 + fr;
            ahi[mi] = *(const v16bf*)&sAhi[cur][r][fk];
            alo[mi] = *(const v16bf*)&sAlo[cur][r][fk];
        }
        #pragma unroll
        for (int ni = 0; ni < 2; ++ni) {
            const int c = nwave * 32 + ni * 16 + fr;
            bhi[ni] = *(const v16bf*)&sBhi[cur][c][fk];
            blo[ni] = *(const v16bf*)&sBlo[cur][c][fk];
        }

        // ---- 12 WMMAs: hi*hi + hi*lo + lo*hi ----
        #pragma unroll
        for (int mi = 0; mi < 2; ++mi)
            #pragma unroll
            for (int ni = 0; ni < 2; ++ni) {
                acc[mi][ni] = wmma_bf16(ahi[mi], bhi[ni], acc[mi][ni]);
                acc[mi][ni] = wmma_bf16(ahi[mi], blo[ni], acc[mi][ni]);
                acc[mi][ni] = wmma_bf16(alo[mi], bhi[ni], acc[mi][ni]);
            }

        if (more) {
            storeA(nxt);      // convert+store next A tile (overlapped with WMMA latency)
            wait_async0();    // next B tile resident in LDS
        }
        __syncthreads();
        cur = nxt;
    }

    // ---- epilogue: C/D layout -> row = 8*(lane>=16)+vgpr, col = lane&15 ----
    const float sg = (mode == 1) ? sigmoidf_(w_global[0]) : 0.0f;
    #pragma unroll
    for (int mi = 0; mi < 2; ++mi) {
        const int rbase = rowA0 + mwave * 32 + mi * 16 + ((lane >> 4) * 8);
        #pragma unroll
        for (int ni = 0; ni < 2; ++ni) {
            const int col = colB0 + nwave * 32 + ni * 16 + (lane & 15);
            #pragma unroll
            for (int r = 0; r < 8; ++r) {
                const int row = rbase + r;
                const float v = acc[mi][ni][r];
                if (mode == 0) {
                    Out[(size_t)row * ldo + col] = v;
                } else {
                    const float s = sg * sigmoidf_(w_local[row]);
                    const float a = fmaxf(v, 0.0f);
                    const float x = Xres[(size_t)row * ldo + col];
                    Out[(size_t)row * ldo + col] = fmaf(s, a - x, x);
                }
            }
        }
    }
}

extern "C" void kernel_launch(void* const* d_in, const int* in_sizes, int n_in,
                              void* d_out, int out_size, void* d_ws, size_t ws_size,
                              hipStream_t stream) {
    (void)in_sizes; (void)n_in; (void)out_size; (void)ws_size;
    const float* X        = (const float*)d_in[0];  // [8192, 256]
    const float* A_hat    = (const float*)d_in[1];  // [8192, 8192]
    const float* w_global = (const float*)d_in[2];  // [1]
    const float* w_local  = (const float*)d_in[3];  // [8192]
    const float* W_agg    = (const float*)d_in[4];  // [256, 256]
    float*       out      = (float*)d_out;

    char* ws = (char*)d_ws;
    float*  T     = (float*)ws;                                    // 8 MB
    __bf16* Xt_hi = (__bf16*)(ws + ((size_t)8 << 20));             // 4 MB  [256][8192]
    __bf16* Xt_lo = (__bf16*)(ws + ((size_t)12 << 20));            // 4 MB
    __bf16* Wt_hi = (__bf16*)(ws + ((size_t)16 << 20));            // 128 KB [256][256]
    __bf16* Wt_lo = (__bf16*)(ws + ((size_t)16 << 20) + (size_t)D_FEAT * D_FEAT * 2);

    // one-time transpose+split of the static B operands
    transpose_split<<<dim3(N_NODES / 32, D_FEAT / 32), 256, 0, stream>>>(
        X, N_NODES, D_FEAT, Xt_hi, Xt_lo);
    transpose_split<<<dim3(D_FEAT / 32, D_FEAT / 32), 256, 0, stream>>>(
        W_agg, D_FEAT, D_FEAT, Wt_hi, Wt_lo);

    dim3 grid(D_FEAT / TILE_N, N_NODES / TILE_M);  // (4, 64); x-fastest keeps the 4
    dim3 block(256);                               // N-blocks of one M-panel L2-coherent

    // T = A_hat @ X  (HBM-bound stream of A_hat; ~11.6 us roofline floor)
    gcn_gemm_bf16x3<<<grid, block, 0, stream>>>(
        A_hat, N_NODES, Xt_hi, Xt_lo, N_NODES, /*mode=*/0,
        nullptr, nullptr, nullptr, T, D_FEAT);

    // out = s*relu(T @ W_agg) + (1-s)*X  (T L2-hot, fused gated epilogue)
    gcn_gemm_bf16x3<<<grid, block, 0, stream>>>(
        T, D_FEAT, Wt_hi, Wt_lo, D_FEAT, /*mode=*/1,
        X, w_local, w_global, out, D_FEAT);
}
